// PatchProjOut_15659450761472
// MI455X (gfx1250) — matrix operand order repaired and checked by reading, not compiled, not run on hardware
//
#include <hip/hip_runtime.h>

// ---------------- types for WMMA / async ----------------
typedef __attribute__((ext_vector_type(16))) __bf16 v16bf;
typedef __attribute__((ext_vector_type(8)))  float  v8f;
typedef __attribute__((ext_vector_type(4)))  int    v4i;

union Frag16 { uint4 q[2]; v16bf v; };

// ---------------- problem constants ----------------
#define BATCH   32
#define NTOK    1024
#define M_TOT   (BATCH * NTOK)   // 32768
#define K_DIM   1024
#define N_DIM   768              // CH*PS*PS = 3*16*16
#define CH      3
#define PS      16
#define NP      32
#define IMG     (NP * PS)        // 512

#define TILE_M  128
#define TILE_N  256
#define TILE_K  32
#define NSTEPS  (K_DIM / TILE_K) // 32
#define LDS_STRIDE 40            // ushorts/row: 32 data + 8 pad (80B, 16B-aligned)

// ---------------- gfx1250 async global->LDS path (guarded) ----------------
#if defined(__has_builtin)
#if __has_builtin(__builtin_amdgcn_global_load_async_to_lds_b128) && \
    __has_builtin(__builtin_amdgcn_s_wait_asynccnt)
#define USE_ASYNC_LDS 1
#endif
#endif

__device__ __forceinline__ void copy16(const unsigned short* g, unsigned short* l) {
#ifdef USE_ASYNC_LDS
  __builtin_amdgcn_global_load_async_to_lds_b128(
      (__attribute__((address_space(1))) v4i*)g,
      (__attribute__((address_space(3))) v4i*)l, 0, 0);
#else
  *(uint4*)l = *(const uint4*)g;
#endif
}

__device__ __forceinline__ void wait_copies() {
#ifdef USE_ASYNC_LDS
  __builtin_amdgcn_s_wait_asynccnt(0);
#endif
}

// float -> bf16, round to nearest even
__device__ __forceinline__ unsigned short f2bf(float f) {
  unsigned u = __float_as_uint(f);
  u += 0x7FFFu + ((u >> 16) & 1u);
  return (unsigned short)(u >> 16);
}

// ---------------- kernel 1: LayerNorm + SiLU -> bf16 ----------------
__global__ __launch_bounds__(256) void ln_silu_bf16(
    const float* __restrict__ x, const float* __restrict__ g,
    const float* __restrict__ be, unsigned short* __restrict__ h) {
  const int row = blockIdx.x;
  const int t   = threadIdx.x;
  const float4* xr = (const float4*)(x + (size_t)row * K_DIM);
  float4 v = xr[t];

  __shared__ float s1[256], s2[256];
  s1[t] = v.x + v.y + v.z + v.w;
  s2[t] = v.x * v.x + v.y * v.y + v.z * v.z + v.w * v.w;
  __syncthreads();
  for (int off = 128; off > 0; off >>= 1) {
    if (t < off) { s1[t] += s1[t + off]; s2[t] += s2[t + off]; }
    __syncthreads();
  }
  const float mean = s1[0] * (1.0f / K_DIM);
  const float var  = s2[0] * (1.0f / K_DIM) - mean * mean;
  const float rstd = rsqrtf(var + 1e-5f);

  const float4 gv = ((const float4*)g)[t];
  const float4 bv = ((const float4*)be)[t];

  float4 hv;
  hv.x = (v.x - mean) * rstd * gv.x + bv.x;
  hv.y = (v.y - mean) * rstd * gv.y + bv.y;
  hv.z = (v.z - mean) * rstd * gv.z + bv.z;
  hv.w = (v.w - mean) * rstd * gv.w + bv.w;
  hv.x = hv.x / (1.0f + __expf(-hv.x));
  hv.y = hv.y / (1.0f + __expf(-hv.y));
  hv.z = hv.z / (1.0f + __expf(-hv.z));
  hv.w = hv.w / (1.0f + __expf(-hv.w));

  ushort4 o;
  o.x = f2bf(hv.x); o.y = f2bf(hv.y); o.z = f2bf(hv.z); o.w = f2bf(hv.w);
  ((ushort4*)(h + (size_t)row * K_DIM))[t] = o;
}

// ---------------- kernel 2: W (f32) -> bf16 ----------------
__global__ __launch_bounds__(256) void conv_w_bf16(
    const float* __restrict__ W, unsigned short* __restrict__ wb) {
  const size_t i = ((size_t)blockIdx.x * 256 + threadIdx.x) * 4;
  float4 v = *(const float4*)(W + i);
  ushort4 o;
  o.x = f2bf(v.x); o.y = f2bf(v.y); o.z = f2bf(v.z); o.w = f2bf(v.w);
  *(ushort4*)(wb + i) = o;
}

// ---------------- kernel 3: bf16 WMMA GEMM + bias + patchify scatter ----------------
// C[M,N] = H[M,K] * W^T, W stored [N,K].
// Workgroup tile 128x256, 8 waves as 2(M) x 4(N); each wave computes 64x64 = 16 WMMA tiles.
__global__ __launch_bounds__(256) void gemm_wmma(
    const unsigned short* __restrict__ hbf,   // [M_TOT, K_DIM] bf16
    const unsigned short* __restrict__ wbf,   // [N_DIM, K_DIM] bf16
    const float* __restrict__ bias,           // [N_DIM]
    float* __restrict__ out) {                // [B, CH, IMG, IMG]
  __shared__ unsigned short lA[2][TILE_M * LDS_STRIDE];   // 2 * 10240 B
  __shared__ unsigned short lB[2][TILE_N * LDS_STRIDE];   // 2 * 20480 B

  const int tid   = threadIdx.x;
  const int lane  = tid & 31;
  const int wid   = tid >> 5;
  const int mOff  = (wid >> 2) * 64;    // 0 or 64
  const int nOff  = (wid & 3) * 64;     // 0,64,128,192
  const int mBase = blockIdx.x * TILE_M;
  const int nBase = blockIdx.y * TILE_N;

  const int laneN  = lane & 15;
  const int laneHi = lane >> 4;

  v8f acc[4][4];
#pragma unroll
  for (int i = 0; i < 4; ++i)
#pragma unroll
    for (int j = 0; j < 4; ++j) acc[i][j] = (v8f)0.0f;

  // cooperative tile copy: A = 512 chunks (16B), B = 1024 chunks; 6 per thread
  auto copy_tiles = [&](int buf, int kk) {
#pragma unroll
    for (int u = 0; u < 2; ++u) {               // A chunks 0..511
      const int chunk = tid + u * 256;
      const int row = chunk >> 2, cr = chunk & 3;
      copy16(hbf + (size_t)(mBase + row) * K_DIM + kk + cr * 8,
             &lA[buf][row * LDS_STRIDE + cr * 8]);
    }
#pragma unroll
    for (int u = 0; u < 4; ++u) {               // B chunks 0..1023
      const int chunk = tid + u * 256;
      const int row = chunk >> 2, cr = chunk & 3;
      copy16(wbf + (size_t)(nBase + row) * K_DIM + kk + cr * 8,
             &lB[buf][row * LDS_STRIDE + cr * 8]);
    }
  };

  copy_tiles(0, 0);
  wait_copies();
  __syncthreads();

  // fragment K-offsets per ISA 7.12.2 (16-bit operands, wave32):
  // A (16x32): lane<16 -> K chunks [0..7],[16..23]; lane>=16 -> [8..15],[24..31]
  // B (32x16): lane = N col; lane<16 -> K chunks [0..7],[16..23]; lane>=16 -> [8..15],[24..31]
  const int aK = laneHi ? 8 : 0;
  const int bK = laneHi ? 8 : 0;

  for (int s = 0; s < NSTEPS; ++s) {
    const int cur = s & 1;
    if (s + 1 < NSTEPS) {
      copy_tiles(cur ^ 1, (s + 1) * TILE_K);
      if (s + 2 < NSTEPS)
        __builtin_prefetch(hbf + (size_t)(mBase + (tid >> 2)) * K_DIM + (s + 2) * TILE_K, 0, 1);
    }

    Frag16 fa[4], fb[4];
#pragma unroll
    for (int i = 0; i < 4; ++i) {
      const unsigned short* p = &lA[cur][(mOff + i * 16 + laneN) * LDS_STRIDE + aK];
      fa[i].q[0] = *(const uint4*)(p);
      fa[i].q[1] = *(const uint4*)(p + 16);
    }
#pragma unroll
    for (int j = 0; j < 4; ++j) {
      const unsigned short* p = &lB[cur][(nOff + j * 16 + laneN) * LDS_STRIDE + bK];
      fb[j].q[0] = *(const uint4*)(p);
      fb[j].q[1] = *(const uint4*)(p + 16);
    }
#pragma unroll
    for (int i = 0; i < 4; ++i)
#pragma unroll
      for (int j = 0; j < 4; ++j)
        acc[i][j] = __builtin_amdgcn_wmma_f32_16x16x32_bf16(
            false, fa[i].v, false, fb[j].v, (short)0, acc[i][j], false, false);

    wait_copies();          // next-buffer async copies done before barrier
    __syncthreads();        // also flushes this wave's ds/global traffic
  }

  // ---- epilogue: bias + patchify scatter  o=(ph*PS+pw)*CH+c ; n=h*NP+w ----
#pragma unroll
  for (int i = 0; i < 4; ++i) {
#pragma unroll
    for (int j = 0; j < 4; ++j) {
      const int col = nBase + nOff + j * 16 + laneN;   // output feature o
      const float bs = bias[col];
      const int c  = col % CH;
      const int pp = col / CH;        // ph*PS + pw
      const int pw = pp & (PS - 1);
      const int ph = pp >> 4;
#pragma unroll
      for (int r = 0; r < 8; ++r) {
        const int mg  = mBase + mOff + i * 16 + laneHi * 8 + r;  // row in [0,M)
        const int bb  = mg >> 10;          // batch
        const int tok = mg & (NTOK - 1);
        const int hh  = tok >> 5;          // patch row
        const int ww  = tok & (NP - 1);    // patch col
        const size_t off =
            (((size_t)(bb * CH + c) * IMG) + (hh * PS + ph)) * IMG + (ww * PS + pw);
        out[off] = acc[i][j][r] + bs;
      }
    }
  }
}

// ---------------- launcher ----------------
extern "C" void kernel_launch(void* const* d_in, const int* in_sizes, int n_in,
                              void* d_out, int out_size, void* d_ws, size_t ws_size,
                              hipStream_t stream) {
  (void)in_sizes; (void)n_in; (void)out_size; (void)ws_size;
  const float* x    = (const float*)d_in[0];
  const float* ln_w = (const float*)d_in[1];
  const float* ln_b = (const float*)d_in[2];
  const float* W    = (const float*)d_in[3];
  const float* bias = (const float*)d_in[4];
  float* out        = (float*)d_out;

  unsigned short* hbf = (unsigned short*)d_ws;                    // 64 MB bf16 activations
  unsigned short* wbf = hbf + (size_t)M_TOT * K_DIM;              // 1.5 MB bf16 weights

  ln_silu_bf16<<<M_TOT, 256, 0, stream>>>(x, ln_w, ln_b, hbf);
  conv_w_bf16<<<(N_DIM * K_DIM) / (256 * 4), 256, 0, stream>>>(W, wbf);

  dim3 grid(M_TOT / TILE_M, N_DIM / TILE_N);                      // 256 x 3
  gemm_wmma<<<grid, 256, 0, stream>>>(hbf, wbf, bias, out);
}
